// MambaBlock_32349693673991
// MI455X (gfx1250) — matrix operand order repaired and checked by reading, not compiled, not run on hardware
//
#include <hip/hip_runtime.h>
#include <hip/hip_bf16.h>
#include <math.h>

// ---- problem dims (from reference) ----
#define D_MODEL   512
#define D_STATE   64
#define D_CONV    4
#define HEADDIM   64
#define D_INNER   1024
#define NHEADS    16
#define CONV_DIM  1152        // D_INNER + 2*D_STATE
#define D_IN_PROJ 2192        // 2*D_INNER + 2*D_STATE + NHEADS
#define BATCH     4
#define SEQ       1024
#define ROWS      (BATCH * SEQ)   // 4096
#define EPS       1e-5f

typedef __bf16 bf16_t;
typedef __attribute__((ext_vector_type(16))) __bf16 v16bf;
typedef __attribute__((ext_vector_type(8)))  __bf16 v8bf;
typedef __attribute__((ext_vector_type(8)))  float  v8f;

// ---------------------------------------------------------------------------
// gfx1250 async global->LDS copy (ASYNCcnt path), with sync fallback.
// Probe-discovered builtin signature (from hipcc diagnostic):
//   void __builtin_amdgcn_global_load_async_to_lds_b128(
//        v4i addrspace(1)* src, v4i addrspace(3)* dst, imm offset, imm cpol)
// where v4i = int __attribute__((vector_size(16))).
// ---------------------------------------------------------------------------
#if defined(__AMDGCN__) && __has_builtin(__builtin_amdgcn_global_load_async_to_lds_b128)
#define ASYNC_LDS 1
#else
#define ASYNC_LDS 0
#endif

typedef int v4i_vs __attribute__((vector_size(16)));
typedef __attribute__((address_space(1))) v4i_vs* gv4i_p;
typedef __attribute__((address_space(3))) v4i_vs* lv4i_p;

__device__ __forceinline__ void async_cp16(const bf16_t* g, bf16_t* l) {
#if ASYNC_LDS
  __builtin_amdgcn_global_load_async_to_lds_b128((gv4i_p)g, (lv4i_p)l, 0, 0);
#else
  *(v8bf*)l = *(const v8bf*)g;
#endif
}

__device__ __forceinline__ void async_wait_all() {
#if ASYNC_LDS
#if __has_builtin(__builtin_amdgcn_s_wait_asynccnt)
  __builtin_amdgcn_s_wait_asynccnt(0);
#else
  asm volatile("s_wait_asynccnt 0x0" ::: "memory");
#endif
#endif
}

// ---------------------------------------------------------------------------
// f32 -> bf16 conversion
// ---------------------------------------------------------------------------
__global__ __launch_bounds__(256)
void k_cvt_bf16(const float* __restrict__ src, bf16_t* __restrict__ dst, int n) {
  int i = blockIdx.x * 256 + threadIdx.x;
  if (i < n) dst[i] = (bf16_t)src[i];
}

// ---------------------------------------------------------------------------
// GEMM: C[M,N] = A[M,K] * B[N,K]^T (row-major, K contiguous), bf16 -> f32.
// Block = 256 threads = 8 waves in a 2(M) x 4(N) grid; each wave computes a
// 2x2 set of 16x16 accumulators (wave tile 32x32, block tile 64M x 128N).
// K-tiles of 32 are staged global->LDS with async-to-LDS DMA, double-buffered
// so the next tile's copy overlaps the current tile's 32 wmmas.
// Fragment layouts as loaded here (A and B share contiguous k per lane half):
//   A 16x32: lane<16 -> row=lane, K 0..15 ; lane>=16 -> row=lane-16, K 16..31
//   B 32x16: lane<16 -> col=lane, K 0..15 ; lane>=16 -> col=lane-16, K 16..31
//   C 16x16: vgpr r -> M = r + 8*(lane>=16), N = lane&15
// N may be ragged (D_IN_PROJ = 2192 = 17.125 * 128): B-row clamp + store guard.
// ---------------------------------------------------------------------------
__global__ __launch_bounds__(256)
void k_gemm_bf16_wmma(const bf16_t* __restrict__ A,
                      const bf16_t* __restrict__ B,
                      float* __restrict__ C,
                      int N, int K) {
  __shared__ __align__(64) bf16_t sA[2][64 * 32];    // 8KB
  __shared__ __align__(64) bf16_t sB[2][128 * 32];   // 16KB

  const int tid  = threadIdx.x;
  const int lane = tid & 31;
  const int wave = tid >> 5;
  const int wm   = wave >> 2;          // 0..1  (M direction)
  const int wn   = wave & 3;           // 0..3  (N direction)
  const int l15  = lane & 15;
  const int hi   = (lane >> 4) & 1;

  const int mblk = blockIdx.x * 64;
  const int nblk = blockIdx.y * 128;

  // async-copy work assignment: A tile = 256 16B chunks (1/thread),
  // B tile = 512 16B chunks (2/thread)
  const int ar = tid >> 2;             // A row 0..63
  const int ac = (tid & 3) * 8;        // A col (elements)

  auto issue_tile = [&](int buf, int k0) {
    const bf16_t* ga = A + (size_t)(mblk + ar) * K + k0 + ac;
    async_cp16(ga, &sA[buf][ar * 32 + ac]);
#pragma unroll
    for (int j = 0; j < 2; ++j) {
      int ch  = tid + j * 256;
      int row = ch >> 2;               // 0..127
      int c   = (ch & 3) * 8;
      int nrow = nblk + row;
      if (nrow >= N) nrow = N - 1;     // clamp ragged edge (result discarded)
      const bf16_t* gb = B + (size_t)nrow * K + k0 + c;
      async_cp16(gb, &sB[buf][row * 32 + c]);
    }
  };

  v8f acc[2][2];
#pragma unroll
  for (int i = 0; i < 2; ++i)
#pragma unroll
    for (int j = 0; j < 2; ++j)
      acc[i][j] = (v8f){0.f, 0.f, 0.f, 0.f, 0.f, 0.f, 0.f, 0.f};

  const int nk = K / 32;
  issue_tile(0, 0);

  for (int kt = 0; kt < nk; ++kt) {
    async_wait_all();        // own DMA done
    __syncthreads();         // everyone's DMA visible; prev compute finished
    if (kt + 1 < nk) issue_tile((kt + 1) & 1, (kt + 1) * 32);

    const int buf = kt & 1;
    union { v16bf v; v8bf h[2]; } a[2];
    v16bf bf[2];
#pragma unroll
    for (int i = 0; i < 2; ++i) {
      const int rowa = (wm * 32 + i * 16 + l15) * 32;
      a[i].v = *(const v16bf*)&sA[buf][rowa + hi * 16];
    }
#pragma unroll
    for (int j = 0; j < 2; ++j) {
      const int rowb = (wn * 32 + j * 16 + l15) * 32;
      bf[j] = *(const v16bf*)&sB[buf][rowb + hi * 16];
    }
#pragma unroll
    for (int i = 0; i < 2; ++i)
#pragma unroll
      for (int j = 0; j < 2; ++j)
        acc[i][j] = __builtin_amdgcn_wmma_f32_16x16x32_bf16(
            false, a[i].v, false, bf[j], (short)0, acc[i][j], false, false);
  }

#pragma unroll
  for (int i = 0; i < 2; ++i) {
    const int mrow = mblk + wm * 32 + i * 16 + hi * 8;
#pragma unroll
    for (int j = 0; j < 2; ++j) {
      const int col = nblk + wn * 32 + j * 16 + l15;
      if (col < N) {
        float* Cp = C + (size_t)mrow * N + col;
#pragma unroll
        for (int r = 0; r < 8; ++r) Cp[(size_t)r * N] = acc[i][j][r];
      }
    }
  }
}

// ---------------------------------------------------------------------------
// Causal depthwise conv1d (kernel=4) + SiLU over the xBC slice of zxbcdt.
// ---------------------------------------------------------------------------
__global__ __launch_bounds__(256)
void k_conv_silu(const float* __restrict__ zx,
                 const float* __restrict__ conv_w,
                 const float* __restrict__ conv_b,
                 float* __restrict__ xc) {
  int i = blockIdx.x * 256 + threadIdx.x;
  if (i >= BATCH * SEQ * CONV_DIM) return;
  int c = i % CONV_DIM;
  int l = (i / CONV_DIM) % SEQ;
  int b = i / (CONV_DIM * SEQ);
  float acc = conv_b[c];
#pragma unroll
  for (int k = 0; k < D_CONV; ++k) {
    int lt = l + k - (D_CONV - 1);
    if (lt >= 0)
      acc += zx[(size_t)(b * SEQ + lt) * D_IN_PROJ + D_INNER + c] * conv_w[c * D_CONV + k];
  }
  acc = acc / (1.f + __expf(-acc));           // silu
  xc[i] = acc;
}

// ---------------------------------------------------------------------------
// dt = softplus(dt_raw + dt_bias); dA = exp(dt * (-exp(A_log)))
// ---------------------------------------------------------------------------
__global__ __launch_bounds__(256)
void k_dt(const float* __restrict__ zx,
          const float* __restrict__ dt_bias,
          const float* __restrict__ A_log,
          float* __restrict__ dt_o, float* __restrict__ dA_o) {
  int i = blockIdx.x * 256 + threadIdx.x;
  if (i >= ROWS * NHEADS) return;
  int h   = i % NHEADS;
  int row = i / NHEADS;
  float d  = zx[(size_t)row * D_IN_PROJ + D_INNER + CONV_DIM + h] + dt_bias[h];
  float sp = (d > 20.f) ? d : log1pf(__expf(d));
  float A  = -__expf(A_log[h]);
  dt_o[i] = sp;
  dA_o[i] = __expf(sp * A);
}

// ---------------------------------------------------------------------------
// Selective scan: one block per (batch, head). 64x64 state in registers
// (16 f32 per thread). Per-chunk staging of x/B/C/dA/dt through LDS;
// y reduction over the state dim via wave32 shfl_xor across 4-lane groups.
// ---------------------------------------------------------------------------
#define TCHUNK 8
__global__ __launch_bounds__(256)
void k_scan(const float* __restrict__ xc,
            const float* __restrict__ dt_b,
            const float* __restrict__ dA_b,
            const float* __restrict__ D_skip,
            float* __restrict__ yraw) {
  __shared__ float sxs[TCHUNK * HEADDIM];
  __shared__ float sB [TCHUNK * D_STATE];
  __shared__ float sC [TCHUNK * D_STATE];
  __shared__ float sdA[TCHUNK];
  __shared__ float sdt[TCHUNK];

  const int bh  = blockIdx.x;             // b*NHEADS + h
  const int b   = bh / NHEADS;
  const int h   = bh % NHEADS;
  const int tid = threadIdx.x;
  const int p   = tid >> 2;               // headdim index 0..63
  const int nq  = tid & 3;                // 16-wide state-column chunk
  const float dsk = D_skip[h];

  float st[16];
#pragma unroll
  for (int j = 0; j < 16; ++j) st[j] = 0.f;

  for (int t0 = 0; t0 < SEQ; t0 += TCHUNK) {
    __syncthreads();
    for (int i = tid; i < TCHUNK * HEADDIM; i += 256) {
      int tt = i >> 6, e = i & 63;
      size_t rowoff = (size_t)(b * SEQ + t0 + tt) * CONV_DIM;
      sxs[i] = xc[rowoff + h * HEADDIM + e];
      sB[i]  = xc[rowoff + D_INNER + e];
      sC[i]  = xc[rowoff + D_INNER + D_STATE + e];
    }
    if (tid < TCHUNK) {
      int idx = (b * SEQ + t0 + tid) * NHEADS + h;
      sdA[tid] = dA_b[idx];
      sdt[tid] = dt_b[idx];
    }
    __syncthreads();

    for (int tt = 0; tt < TCHUNK; ++tt) {
      float dA   = sdA[tt];
      float dtv  = sdt[tt];
      float xv   = sxs[tt * HEADDIM + p];
      float coef = dtv * xv;
      float y = 0.f;
#pragma unroll
      for (int j = 0; j < 16; ++j) {
        float Bv = sB[tt * D_STATE + nq * 16 + j];
        float Cv = sC[tt * D_STATE + nq * 16 + j];
        st[j] = st[j] * dA + coef * Bv;
        y += st[j] * Cv;
      }
      y += __shfl_xor(y, 1, 32);
      y += __shfl_xor(y, 2, 32);
      if (nq == 0)
        yraw[(size_t)(b * SEQ + t0 + tt) * D_INNER + h * HEADDIM + p] = y + dsk * xv;
    }
  }
}

// ---------------------------------------------------------------------------
// y *= silu(z); y = y * rsqrt(mean(y^2)+eps) * norm_w;  cast to bf16.
// ---------------------------------------------------------------------------
__global__ __launch_bounds__(256)
void k_gate_rms(const float* __restrict__ zx,
                const float* __restrict__ yraw,
                const float* __restrict__ norm_w,
                bf16_t* __restrict__ ybf) {
  __shared__ float red[256];
  const int row = blockIdx.x;
  const int tid = threadIdx.x;
  float yv[4];
  float ss = 0.f;
#pragma unroll
  for (int i = 0; i < 4; ++i) {
    int j = tid + i * 256;
    float z = zx[(size_t)row * D_IN_PROJ + j];
    float y = yraw[(size_t)row * D_INNER + j] * (z / (1.f + __expf(-z)));
    yv[i] = y;
    ss += y * y;
  }
  red[tid] = ss;
  __syncthreads();
  for (int s = 128; s > 0; s >>= 1) {
    if (tid < s) red[tid] += red[tid + s];
    __syncthreads();
  }
  float rms = rsqrtf(red[0] / (float)D_INNER + EPS);
#pragma unroll
  for (int i = 0; i < 4; ++i) {
    int j = tid + i * 256;
    ybf[(size_t)row * D_INNER + j] = (bf16_t)(yv[i] * rms * norm_w[j]);
  }
}

// ---------------------------------------------------------------------------
// out = layernorm(mout + x) * ln_w + ln_b.  One block per row of 512.
// ---------------------------------------------------------------------------
__global__ __launch_bounds__(256)
void k_resid_ln(const float* __restrict__ mout,
                const float* __restrict__ x,
                const float* __restrict__ lw,
                const float* __restrict__ lb,
                float* __restrict__ out) {
  __shared__ float s1[256], s2[256];
  const int row = blockIdx.x;
  const int tid = threadIdx.x;
  float v[2];
  float sum = 0.f, sq = 0.f;
#pragma unroll
  for (int i = 0; i < 2; ++i) {
    int j = tid + i * 256;
    float t = mout[(size_t)row * D_MODEL + j] + x[(size_t)row * D_MODEL + j];
    v[i] = t;
    sum += t;
    sq  += t * t;
  }
  s1[tid] = sum; s2[tid] = sq;
  __syncthreads();
  for (int s = 128; s > 0; s >>= 1) {
    if (tid < s) { s1[tid] += s1[tid + s]; s2[tid] += s2[tid + s]; }
    __syncthreads();
  }
  float mu  = s1[0] / (float)D_MODEL;
  float var = s2[0] / (float)D_MODEL - mu * mu;
  float inv = rsqrtf(var + EPS);
#pragma unroll
  for (int i = 0; i < 2; ++i) {
    int j = tid + i * 256;
    out[(size_t)row * D_MODEL + j] = (v[i] - mu) * inv * lw[j] + lb[j];
  }
}

// ---------------------------------------------------------------------------
extern "C" void kernel_launch(void* const* d_in, const int* in_sizes, int n_in,
                              void* d_out, int out_size, void* d_ws, size_t ws_size,
                              hipStream_t stream) {
  (void)in_sizes; (void)n_in; (void)out_size; (void)ws_size;
  const float* x       = (const float*)d_in[0];
  const float* in_w    = (const float*)d_in[1];   // (2192, 512)
  const float* conv_w  = (const float*)d_in[2];   // (1152, 4)
  const float* conv_b  = (const float*)d_in[3];
  const float* dt_bias = (const float*)d_in[4];
  const float* A_log   = (const float*)d_in[5];
  const float* D_skip  = (const float*)d_in[6];
  const float* norm_w  = (const float*)d_in[7];
  const float* out_w   = (const float*)d_in[8];   // (512, 1024)
  const float* ln_w    = (const float*)d_in[9];
  const float* ln_b    = (const float*)d_in[10];
  float* out = (float*)d_out;

  // workspace carving (256B aligned)
  char* ws = (char*)d_ws;
  size_t off = 0;
  auto carve = [&](size_t bytes) {
    char* p = ws + off;
    off = (off + bytes + 255) & ~(size_t)255;
    return p;
  };
  bf16_t* xbf  = (bf16_t*)carve((size_t)ROWS * D_MODEL * 2);
  bf16_t* wibf = (bf16_t*)carve((size_t)D_IN_PROJ * D_MODEL * 2);
  bf16_t* wobf = (bf16_t*)carve((size_t)D_MODEL * D_INNER * 2);
  float*  zx   = (float*)carve((size_t)ROWS * D_IN_PROJ * 4);
  float*  xc   = (float*)carve((size_t)ROWS * CONV_DIM * 4);
  float*  dtb  = (float*)carve((size_t)ROWS * NHEADS * 4);
  float*  dab  = (float*)carve((size_t)ROWS * NHEADS * 4);
  float*  yraw = (float*)carve((size_t)ROWS * D_INNER * 4);
  bf16_t* ybf  = (bf16_t*)carve((size_t)ROWS * D_INNER * 2);
  float*  mout = (float*)carve((size_t)ROWS * D_MODEL * 4);

  // 1) bf16 conversions for the two GEMMs
  k_cvt_bf16<<<(ROWS * D_MODEL + 255) / 256, 256, 0, stream>>>(x, xbf, ROWS * D_MODEL);
  k_cvt_bf16<<<(D_IN_PROJ * D_MODEL + 255) / 256, 256, 0, stream>>>(in_w, wibf, D_IN_PROJ * D_MODEL);
  k_cvt_bf16<<<(D_MODEL * D_INNER + 255) / 256, 256, 0, stream>>>(out_w, wobf, D_MODEL * D_INNER);

  // 2) in_proj: (4096,512) x (2192,512)^T -> (4096,2192); ragged N edge
  k_gemm_bf16_wmma<<<dim3(ROWS / 64, (D_IN_PROJ + 127) / 128), 256, 0, stream>>>(
      xbf, wibf, zx, D_IN_PROJ, D_MODEL);

  // 3) causal depthwise conv + silu on xBC slice
  k_conv_silu<<<(BATCH * SEQ * CONV_DIM + 255) / 256, 256, 0, stream>>>(zx, conv_w, conv_b, xc);

  // 4) dt softplus + dA
  k_dt<<<(ROWS * NHEADS + 255) / 256, 256, 0, stream>>>(zx, dt_bias, A_log, dtb, dab);

  // 5) selective scan (64 blocks = batch*heads)
  k_scan<<<BATCH * NHEADS, 256, 0, stream>>>(xc, dtb, dab, D_skip, yraw);

  // 6) gate with silu(z), RMSNorm, cast bf16
  k_gate_rms<<<ROWS, 256, 0, stream>>>(zx, yraw, norm_w, ybf);

  // 7) out_proj: (4096,1024) x (512,1024)^T -> (4096,512)
  k_gemm_bf16_wmma<<<dim3(ROWS / 64, D_MODEL / 128), 256, 0, stream>>>(
      ybf, wobf, mout, D_MODEL, D_INNER);

  // 8) residual + layernorm -> out
  k_resid_ln<<<ROWS, 256, 0, stream>>>(mout, x, ln_w, ln_b, out);
}
